// MultiHeadAttention__24919400252075
// MI455X (gfx1250) — compile-verified, matrix operand order checked by reading, not run on hardware
//
#include <hip/hip_runtime.h>
#include <hip/hip_bf16.h>

typedef __attribute__((ext_vector_type(16))) _Float16 v16h;
typedef __attribute__((ext_vector_type(8)))  _Float16 v8h;
typedef __attribute__((ext_vector_type(8)))  float    v8f;
typedef unsigned int u32x4 __attribute__((ext_vector_type(4)));
typedef int          i32x8 __attribute__((ext_vector_type(8)));
typedef int          i32x4 __attribute__((ext_vector_type(4)));

#define B_  2
#define N_  2048
#define D_  1024
#define H_  16
#define DH_ 64
#define M_  (B_ * N_)   // 4096 rows total

// ---------------------------------------------------------------------------
// WMMA fragment helpers (CDNA5 16x16x32 f16 layouts, wave32)
// A frag: row m = lane%16, halves k = (h<8?h:h+8) + 8*(lane/16)
// B frag: same pattern applied to rows of B^T (lane = output column n)
// C frag: row = r + 8*(lane/16), col = lane%16
// ---------------------------------------------------------------------------
__device__ __forceinline__ v16h ld_frag16(const _Float16* p) {
  v8h lo = *(const v8h*)(p);
  v8h hi = *(const v8h*)(p + 16);
  v16h r;
#pragma unroll
  for (int i = 0; i < 8; ++i) { r[i] = lo[i]; r[i + 8] = hi[i]; }
  return r;
}

__device__ __forceinline__ v16h frag_rows(const _Float16* base, int row0,
                                          int stride, int kbase) {
  const int lane = threadIdx.x & 31;
  const _Float16* p = base + (size_t)(row0 + (lane & 15)) * stride + kbase +
                      ((lane >> 4) << 3);
  return ld_frag16(p);
}

__device__ __forceinline__ v8f wmma_f16(v16h a, v16h b, v8f c) {
  return __builtin_amdgcn_wmma_f32_16x16x32_f16(false, a, false, b,
                                                (short)0, c, false, false);
}

// ---------------------------------------------------------------------------
// Tensor Data Mover: 2D tile (f16 elements) global -> LDS.
// Descriptor per CDNA5 ISA:
//  group0: [0]=count(1), [1]=lds_addr, [2]=gaddr[31:0],
//          [3]=gaddr[56:32] | type(2)<<30
//  group1: [0]=data_size(1=2B)<<16, tensor_dim0 @bits79:48,
//          tensor_dim1 @111:80, tile_dim0 @127:112, tile_dim1 @143:128,
//          tensor_dim0_stride @207:160
// Tracked by TENSORcnt; issue from one wave only (EXEC ignored).
// This toolchain exposes the 6-arg builtin: (g0, g1, g2, g3, g4, cpol).
// ---------------------------------------------------------------------------
__device__ __forceinline__ void tdm_load_2d(unsigned int lds_addr,
                                            const _Float16* gptr,
                                            unsigned int tensor0,
                                            unsigned int tensor1,
                                            unsigned int stride0,
                                            unsigned int tile0,
                                            unsigned int tile1) {
  unsigned long long ga = (unsigned long long)(size_t)gptr;
  u32x4 g0;
  g0[0] = 1u;                                        // count = 1 (valid)
  g0[1] = lds_addr;                                  // LDS byte address
  g0[2] = (unsigned int)ga;                          // global addr lo
  g0[3] = (unsigned int)((ga >> 32) & 0x01FFFFFFu) | (2u << 30);  // hi|type=2
  i32x8 g1;
  g1[0] = (int)(1u << 16);                           // data_size = 2 bytes
  g1[1] = (int)((tensor0 & 0xFFFFu) << 16);          // tensor_dim0 lo16
  g1[2] = (int)((tensor0 >> 16) | ((tensor1 & 0xFFFFu) << 16));
  g1[3] = (int)((tensor1 >> 16) | (tile0 << 16));    // tile_dim0
  g1[4] = (int)(tile1 & 0xFFFFu);                    // tile_dim1 (tile_dim2=0)
  g1[5] = (int)stride0;                              // dim0 stride lo32
  g1[6] = 0;
  g1[7] = 0;
  i32x4 z4 = {0, 0, 0, 0};
  i32x8 z8 = {0, 0, 0, 0, 0, 0, 0, 0};
  __builtin_amdgcn_tensor_load_to_lds(g0, g1, z4, z4, z8, 0);
}

__device__ __forceinline__ unsigned int lds_off(const void* p) {
  return (unsigned int)(size_t)p;  // generic LDS ptr: offset lives in [31:0]
}

// ---------------------------------------------------------------------------
// Stage 1: fp32 -> fp16 convert (activations) and convert+transpose (weights)
// ---------------------------------------------------------------------------
__global__ void cvt_f32_f16(const float* __restrict__ in,
                            _Float16* __restrict__ out, int n) {
  int i = blockIdx.x * blockDim.x + threadIdx.x;
  if (i < n) out[i] = (_Float16)in[i];
}

__global__ void transpose_cvt(const float* __restrict__ W,
                              _Float16* __restrict__ Wt) {
  int t = blockIdx.x * blockDim.x + threadIdx.x;
  int j = t >> 10;
  int i = t & (D_ - 1);
  Wt[(size_t)j * D_ + i] = (_Float16)W[(size_t)i * D_ + j];
}

// ---------------------------------------------------------------------------
// Stage 2: projection GEMM  Y = (X @ W + bias) * scale, head-split f16 output.
// B-tile (64 cols x 32 k) is shared by all 8 waves -> staged via TDM into
// double-buffered LDS, prefetching k+32 while computing k.
// ---------------------------------------------------------------------------
__global__ void gemm_proj(const _Float16* __restrict__ X,
                          const _Float16* __restrict__ Wt,
                          const float* __restrict__ bias,
                          _Float16* __restrict__ Y,
                          float scale, int transV) {
  __shared__ alignas(16) _Float16 Bsh[2][64][32];

  const int wave = threadIdx.x >> 5;
  const int lane = threadIdx.x & 31;
  const int row0 = blockIdx.x * 128 + wave * 16;
  const int col0 = blockIdx.y * 64;
  const _Float16* Wtile = Wt + (size_t)col0 * D_;

  if (wave == 0)
    tdm_load_2d(lds_off(&Bsh[0][0][0]), Wtile, D_, D_, D_, 32, 64);

  v8f acc[4] = {};
  for (int kc = 0; kc < D_ / 32; ++kc) {
    const int buf = kc & 1;
    if (wave == 0) __builtin_amdgcn_s_wait_tensorcnt(0);
    __syncthreads();  // buf ready; everyone done reading buf^1
    if (wave == 0 && kc + 1 < D_ / 32)
      tdm_load_2d(lds_off(&Bsh[buf ^ 1][0][0]), Wtile + (kc + 1) * 32,
                  D_, D_, D_, 32, 64);

    v16h a = frag_rows(X, row0, D_, kc * 32);
#pragma unroll
    for (int t = 0; t < 4; ++t) {
      v16h b = frag_rows(&Bsh[buf][0][0], t * 16, 32, 0);
      acc[t] = wmma_f16(a, b, acc[t]);
    }
  }

  const int half = lane >> 4;
  const int n = lane & 15;
#pragma unroll
  for (int t = 0; t < 4; ++t) {
    const int j = col0 + t * 16 + n;
    const int hh = j >> 6;
    const int dd = j & 63;
    const float bj = bias[j];
#pragma unroll
    for (int r = 0; r < 8; ++r) {
      const int m = row0 + r + 8 * half;
      const int bb = m >> 11;
      const int nn = m & (N_ - 1);
      const float v = (acc[t][r] + bj) * scale;
      size_t idx = transV
          ? (((size_t)(bb * H_ + hh) * DH_ + dd) * N_ + nn)
          : (((size_t)(bb * H_ + hh) * N_ + nn) * DH_ + dd);
      Y[idx] = (_Float16)v;
    }
  }
}

// ---------------------------------------------------------------------------
// Stage 3: causal flash attention with TDM-staged K/V chunks.
// grid (B*H, N/128), block 256 (8 waves); wave owns 16 query rows.
// All waves run the block-uniform chunk count (fully-masked chunks are
// numerically inert: exp->0, scale=1), enabling barriers + shared staging.
// ---------------------------------------------------------------------------
__global__ void flash_attn(const _Float16* __restrict__ Q,
                           const _Float16* __restrict__ K,
                           const _Float16* __restrict__ Vt,
                           _Float16* __restrict__ O) {
  __shared__ alignas(16) _Float16 Ksh[2][32][64];   // 32 keys x 64 dh
  __shared__ alignas(16) _Float16 Vsh[2][64][32];   // 64 dh x 32 keys
  __shared__ alignas(16) _Float16 pbuf[8][16][32];  // per-wave P re-layout

  const int wave = threadIdx.x >> 5;
  const int lane = threadIdx.x & 31;
  const int half = lane >> 4;
  const int n = lane & 15;
  const int bh = blockIdx.x;
  const int b = bh >> 4;
  const int h = bh & 15;
  const int qi0 = blockIdx.y * 128 + wave * 16;

  const _Float16* Qb = Q  + (size_t)bh * N_ * DH_;
  const _Float16* Kb = K  + (size_t)bh * N_ * DH_;
  const _Float16* Vb = Vt + (size_t)bh * DH_ * N_;

  v16h qa[2];
#pragma unroll
  for (int c = 0; c < 2; ++c) qa[c] = frag_rows(Qb, qi0, DH_, c * 32);

  v8f o[4] = {};
  float mrun[8], lrun[8];
#pragma unroll
  for (int r = 0; r < 8; ++r) { mrun[r] = -3.0e38f; lrun[r] = 0.0f; }

  // block-uniform causal bound: keys [0, blk_last_row + 16)
  const int nchunks = ((int)blockIdx.y * 128 + 159) >> 5;

  if (wave == 0) {
    tdm_load_2d(lds_off(&Ksh[0][0][0]), Kb, DH_, N_, DH_, 64, 32);
    tdm_load_2d(lds_off(&Vsh[0][0][0]), Vb, N_, DH_, N_, 32, 64);
  }

  for (int jc = 0; jc < nchunks; ++jc) {
    const int buf = jc & 1;
    if (wave == 0) __builtin_amdgcn_s_wait_tensorcnt(0);
    __syncthreads();  // chunk jc staged; all reads of buf^1 retired
    if (wave == 0 && jc + 1 < nchunks) {
      tdm_load_2d(lds_off(&Ksh[buf ^ 1][0][0]),
                  Kb + (size_t)(jc + 1) * 32 * DH_, DH_, N_, DH_, 64, 32);
      tdm_load_2d(lds_off(&Vsh[buf ^ 1][0][0]),
                  Vb + (jc + 1) * 32, N_, DH_, N_, 32, 64);
    }

    // S = Q @ K^T (two 16-key column tiles, K-dim = DH in 2 steps)
    v8f s[2] = {};
#pragma unroll
    for (int c = 0; c < 2; ++c) {
      v16h kb0 = frag_rows(&Ksh[buf][0][0], 0,  DH_, c * 32);
      v16h kb1 = frag_rows(&Ksh[buf][0][0], 16, DH_, c * 32);
      s[0] = wmma_f16(qa[c], kb0, s[0]);
      s[1] = wmma_f16(qa[c], kb1, s[1]);
    }

    // causal mask + online softmax
    float p0[8], p1[8];
#pragma unroll
    for (int r = 0; r < 8; ++r) {
      const int i  = qi0 + r + 8 * half;
      const int j0 = jc * 32 + n;
      const int j1 = j0 + 16;
      float s0 = (j0 > i) ? -3.0e38f : s[0][r];
      float s1 = (j1 > i) ? -3.0e38f : s[1][r];
      float mx = fmaxf(s0, s1);
#pragma unroll
      for (int msk = 1; msk < 16; msk <<= 1)
        mx = fmaxf(mx, __shfl_xor(mx, msk, 32));
      const float mnew = fmaxf(mrun[r], mx);
      const float sc = __expf(mrun[r] - mnew);
      const float e0 = __expf(s0 - mnew);
      const float e1 = __expf(s1 - mnew);
      float rs = e0 + e1;
#pragma unroll
      for (int msk = 1; msk < 16; msk <<= 1)
        rs += __shfl_xor(rs, msk, 32);
      lrun[r] = lrun[r] * sc + rs;
      mrun[r] = mnew;
#pragma unroll
      for (int t = 0; t < 4; ++t) o[t][r] *= sc;
      p0[r] = e0;
      p1[r] = e1;
    }

    // C-fragment -> A-fragment re-layout of P via wave-private LDS
#pragma unroll
    for (int r = 0; r < 8; ++r) {
      pbuf[wave][r + 8 * half][n]      = (_Float16)p0[r];
      pbuf[wave][r + 8 * half][n + 16] = (_Float16)p1[r];
    }
    asm volatile("s_wait_dscnt 0" ::: "memory");
    const int mrow = lane & 15;
    const int kb = (lane >> 4) << 3;
    v8h plo = *(const v8h*)&pbuf[wave][mrow][kb];
    v8h phi = *(const v8h*)&pbuf[wave][mrow][kb + 16];
    v16h pa;
#pragma unroll
    for (int i2 = 0; i2 < 8; ++i2) { pa[i2] = plo[i2]; pa[i2 + 8] = phi[i2]; }
    asm volatile("" ::: "memory");

    // O += P @ V
#pragma unroll
    for (int t = 0; t < 4; ++t) {
      v16h vb = frag_rows(&Vsh[buf][0][0], t * 16, 32, 0);
      o[t] = wmma_f16(pa, vb, o[t]);
    }
  }

#pragma unroll
  for (int t = 0; t < 4; ++t) {
#pragma unroll
    for (int r = 0; r < 8; ++r) {
      const int i = qi0 + r + 8 * half;
      const int col = h * DH_ + t * 16 + n;
      const float v = o[t][r] / lrun[r];
      O[((size_t)(b * N_ + i)) * D_ + col] = (_Float16)v;
    }
  }
}

// ---------------------------------------------------------------------------
// Stage 4: output GEMM  out = O @ Wo + bo (f32), TDM-staged B tile.
// ---------------------------------------------------------------------------
__global__ void gemm_out(const _Float16* __restrict__ X,
                         const _Float16* __restrict__ Wt,
                         const float* __restrict__ bias,
                         float* __restrict__ Y) {
  __shared__ alignas(16) _Float16 Bsh[2][64][32];

  const int wave = threadIdx.x >> 5;
  const int lane = threadIdx.x & 31;
  const int row0 = blockIdx.x * 128 + wave * 16;
  const int col0 = blockIdx.y * 64;
  const _Float16* Wtile = Wt + (size_t)col0 * D_;

  if (wave == 0)
    tdm_load_2d(lds_off(&Bsh[0][0][0]), Wtile, D_, D_, D_, 32, 64);

  v8f acc[4] = {};
  for (int kc = 0; kc < D_ / 32; ++kc) {
    const int buf = kc & 1;
    if (wave == 0) __builtin_amdgcn_s_wait_tensorcnt(0);
    __syncthreads();
    if (wave == 0 && kc + 1 < D_ / 32)
      tdm_load_2d(lds_off(&Bsh[buf ^ 1][0][0]), Wtile + (kc + 1) * 32,
                  D_, D_, D_, 32, 64);

    v16h a = frag_rows(X, row0, D_, kc * 32);
#pragma unroll
    for (int t = 0; t < 4; ++t) {
      v16h b = frag_rows(&Bsh[buf][0][0], t * 16, 32, 0);
      acc[t] = wmma_f16(a, b, acc[t]);
    }
  }

  const int half = lane >> 4;
  const int n = lane & 15;
#pragma unroll
  for (int t = 0; t < 4; ++t) {
    const int j = col0 + t * 16 + n;
    const float bj = bias[j];
#pragma unroll
    for (int r = 0; r < 8; ++r) {
      const int m = row0 + r + 8 * half;
      Y[(size_t)m * D_ + j] = acc[t][r] + bj;
    }
  }
}

// ---------------------------------------------------------------------------
// Host launcher
// ---------------------------------------------------------------------------
extern "C" void kernel_launch(void* const* d_in, const int* in_sizes, int n_in,
                              void* d_out, int out_size, void* d_ws,
                              size_t ws_size, hipStream_t stream) {
  const float* x_q  = (const float*)d_in[0];
  const float* x_kv = (const float*)d_in[1];
  const float* Wq   = (const float*)d_in[2];
  const float* bq   = (const float*)d_in[3];
  const float* Wk   = (const float*)d_in[4];
  const float* bk   = (const float*)d_in[5];
  const float* Wv   = (const float*)d_in[6];
  const float* bv   = (const float*)d_in[7];
  const float* Wo   = (const float*)d_in[8];
  const float* bo   = (const float*)d_in[9];
  float* out = (float*)d_out;

  char* ws = (char*)d_ws;
  size_t off = 0;
  auto carve = [&](size_t bytes) {
    char* p = ws + off;
    off += bytes;
    return p;
  };
  const size_t xbytes = (size_t)M_ * D_ * sizeof(_Float16);   // 8 MB
  const size_t wbytes = (size_t)D_ * D_ * sizeof(_Float16);   // 2 MB
  _Float16* xq16  = (_Float16*)carve(xbytes);
  _Float16* xkv16 = (_Float16*)carve(xbytes);
  _Float16* wqT   = (_Float16*)carve(wbytes);
  _Float16* wkT   = (_Float16*)carve(wbytes);
  _Float16* wvT   = (_Float16*)carve(wbytes);
  _Float16* woT   = (_Float16*)carve(wbytes);
  _Float16* q16   = (_Float16*)carve(xbytes);   // (B,H,N,DH)
  _Float16* k16   = (_Float16*)carve(xbytes);   // (B,H,N,DH)
  _Float16* vt16  = (_Float16*)carve(xbytes);   // (B,H,DH,N)
  _Float16* o16   = (_Float16*)carve(xbytes);   // (B,N,D)

  const int nelem = M_ * D_;

  cvt_f32_f16<<<nelem / 256, 256, 0, stream>>>(x_q,  xq16,  nelem);
  cvt_f32_f16<<<nelem / 256, 256, 0, stream>>>(x_kv, xkv16, nelem);
  transpose_cvt<<<(D_ * D_) / 256, 256, 0, stream>>>(Wq, wqT);
  transpose_cvt<<<(D_ * D_) / 256, 256, 0, stream>>>(Wk, wkT);
  transpose_cvt<<<(D_ * D_) / 256, 256, 0, stream>>>(Wv, wvT);
  transpose_cvt<<<(D_ * D_) / 256, 256, 0, stream>>>(Wo, woT);

  dim3 ggrid(M_ / 128, D_ / 64);
  gemm_proj<<<ggrid, 256, 0, stream>>>(xq16,  wqT, bq, q16,  0.125f, 0);
  gemm_proj<<<ggrid, 256, 0, stream>>>(xkv16, wkT, bk, k16,  1.0f,   0);
  gemm_proj<<<ggrid, 256, 0, stream>>>(xkv16, wvT, bv, vt16, 1.0f,   1);

  flash_attn<<<dim3(B_ * H_, N_ / 128), 256, 0, stream>>>(q16, k16, vt16, o16);

  gemm_out<<<ggrid, 256, 0, stream>>>(o16, woT, bo, out);
}
